// ODAttention_45140106281295
// MI455X (gfx1250) — matrix operand order repaired and checked by reading, not compiled, not run on hardware
//
#include <hip/hip_runtime.h>

// ---------------------------------------------------------------------------
// ODAttention for MI455X (gfx1250, wave32).
// N=128, FDIM=32, PROJ=128, B*T=128.
// Heavy GEMMs run on v_wmma_f32_16x16x32_bf16 (fp32 accum); sparse sandwich
// exploits the 4-row masks exactly. bf16 conversion uses the native cast
// so the backend can emit v_cvt_pk_bf16_f32; output zero-fill is non-temporal
// (native ext-vector type) so 268 MB of zeros don't evict M / gates from L2.
// ---------------------------------------------------------------------------

typedef unsigned short u16;
typedef __attribute__((ext_vector_type(8)))  unsigned short ush8;
typedef __attribute__((ext_vector_type(16))) unsigned short ush16;
typedef __attribute__((ext_vector_type(16))) __bf16        v16bf;
typedef __attribute__((ext_vector_type(8)))  float         v8f;
typedef __attribute__((ext_vector_type(4)))  float         f32x4;   // clang vector (nontemporal-ok)

static __device__ __forceinline__ u16 f2bf(float x) {
    __bf16 h = (__bf16)x;                    // RNE fptrunc; backend may fuse pairs
    return __builtin_bit_cast(u16, h);       // into v_cvt_pk_bf16_f32
}

static __device__ __forceinline__ v16bf pack_frag(ush8 lo, ush8 hi) {
    ush16 s = __builtin_shufflevector(lo, hi, 0,1,2,3,4,5,6,7,8,9,10,11,12,13,14,15);
    return __builtin_bit_cast(v16bf, s);
}

static __device__ __forceinline__ float wred_max(float v) {
    #pragma unroll
    for (int m = 16; m >= 1; m >>= 1) v = fmaxf(v, __shfl_xor(v, m, 32));
    return v;
}
static __device__ __forceinline__ float wred_sum(float v) {
    #pragma unroll
    for (int m = 16; m >= 1; m >>= 1) v += __shfl_xor(v, m, 32);
    return v;
}

// ---------------------------------------------------------------------------
// fp32 -> bf16 conversion (weights)
// ---------------------------------------------------------------------------
__global__ void k_cvt(const float* __restrict__ src, u16* __restrict__ dst, int n) {
    for (int i = blockIdx.x * blockDim.x + threadIdx.x; i < n; i += gridDim.x * blockDim.x)
        dst[i] = f2bf(src[i]);
}

// ---------------------------------------------------------------------------
// Projection GEMM: X_proj[b,i,p] = sum_k X[b,i,k] * W[p,k] + bias[p]
// branch 0: X[i,k] = M[b, i, k/32, k%32]   (contiguous: i*4096 + k)
// branch 1: Y[i,k] = M[b, k/32, i, k%32]   (contiguous chunk: (k/32)*4096 + i*32 + k%32)
// Block = one (b, branch). 8 waves; wave w owns row tile w, all 8 col tiles.
// ---------------------------------------------------------------------------
__global__ __launch_bounds__(256) void k_proj(const float* __restrict__ M,
                                              const u16*   __restrict__ Wbf,
                                              const float* __restrict__ bo,
                                              const float* __restrict__ bd,
                                              u16*         __restrict__ Xp) {
    const int b = blockIdx.x, br = blockIdx.y;
    const float* Msrc = M + (size_t)b * 524288;
    const u16*   W    = Wbf + (size_t)br * 524288;
    const float* bias = br ? bd : bo;
    u16*         out  = Xp + ((size_t)(br * 128 + b)) * 16384;

    __shared__ u16 Ab[128 * 32];   // A panel [row][k] bf16
    __shared__ u16 Bb[128 * 32];   // B panel transposed [n][k] bf16

    const int tid = threadIdx.x;
    const int wave = tid >> 5, lane = tid & 31, lrow = lane & 15, lhalf = lane >> 4;
    const int m0 = wave * 16;
    const int r = tid >> 1, kh = (tid & 1) * 16;   // staging map: 16 elems/thread

    v8f acc[8] = {};

    for (int kk = 0; kk < 4096; kk += 32) {
        // ---- stage A (fp32 load + cvt) ----
        size_t base = (br == 0) ? ((size_t)r * 4096 + kk + kh)
                                : ((size_t)(kk >> 5) * 4096 + (size_t)r * 32 + kh);
        const f32x4* s4 = (const f32x4*)(Msrc + base);
        f32x4 f0 = s4[0], f1 = s4[1], f2 = s4[2], f3 = s4[3];
        float va[16] = {f0.x,f0.y,f0.z,f0.w, f1.x,f1.y,f1.z,f1.w,
                        f2.x,f2.y,f2.z,f2.w, f3.x,f3.y,f3.z,f3.w};
        u16* ad = &Ab[r * 32 + kh];
        #pragma unroll
        for (int j = 0; j < 16; ++j) ad[j] = f2bf(va[j]);
        // ---- stage B (bf16 copy; B[k,n] = W[n, kk+k], stored transposed) ----
        const ush8* w8 = (const ush8*)(W + (size_t)r * 4096 + kk + kh);
        *(ush8*)&Bb[r * 32 + kh]     = w8[0];
        *(ush8*)&Bb[r * 32 + kh + 8] = w8[1];
        __syncthreads();

        // ---- A fragment (ISA 16-bit A layout: chunks at 8*lhalf, 16+8*lhalf) ----
        const int arow = m0 + lrow;
        v16bf afr = pack_frag(*(const ush8*)&Ab[arow * 32 + 8 * lhalf],
                              *(const ush8*)&Ab[arow * 32 + 16 + 8 * lhalf]);
        #pragma unroll
        for (int ct = 0; ct < 8; ++ct) {
            const int n = ct * 16 + lrow;
            v16bf bfr = pack_frag(*(const ush8*)&Bb[n * 32 + 8 * lhalf],
                                  *(const ush8*)&Bb[n * 32 + 16 + 8 * lhalf]);
            acc[ct] = __builtin_amdgcn_wmma_f32_16x16x32_bf16(
                false, afr, false, bfr, (short)0, acc[ct], false, false);
        }
        __syncthreads();
    }

    // ---- epilogue: + bias, store bf16 (C layout: reg rr -> row rr + 8*lhalf) ----
    #pragma unroll
    for (int ct = 0; ct < 8; ++ct) {
        const int n = ct * 16 + lrow;
        const float bv = bias[n];
        #pragma unroll
        for (int rr = 0; rr < 8; ++rr) {
            const int m = m0 + rr + 8 * lhalf;
            out[(size_t)m * 128 + n] = f2bf(acc[ct][rr] + bv);
        }
    }
}

// ---------------------------------------------------------------------------
// Score GEMM: raw[b,i,j] = sum_d Xp[b,i,d] * W1[i,j,d] + bias2[i,j]
// Block = one (i, branch); rows = batch b, cols = j, K = d = 128.
// ---------------------------------------------------------------------------
__global__ __launch_bounds__(256) void k_scores(const u16*   __restrict__ Xp,
                                                const u16*   __restrict__ W1bf,
                                                const float* __restrict__ b_om,
                                                const float* __restrict__ b_de,
                                                float*       __restrict__ G) {
    const int i = blockIdx.x, br = blockIdx.y;
    const u16* A  = Xp + (size_t)br * 2097152;                      // [b][i][d]
    const u16* Bm = W1bf + ((size_t)br * 128 + i) * 16384;          // [j][d]
    const float* bias2 = (br ? b_de : b_om) + (size_t)i * 128;
    float* out = G + (size_t)br * 2097152;                          // [b][i][j]

    const int tid = threadIdx.x;
    const int wave = tid >> 5, lane = tid & 31, lrow = lane & 15, lhalf = lane >> 4;
    const int bb0 = wave * 16;

    v8f acc[8] = {};
    #pragma unroll
    for (int kk = 0; kk < 128; kk += 32) {
        const u16* arp = A + (size_t)(bb0 + lrow) * 16384 + (size_t)i * 128;
        v16bf afr = pack_frag(*(const ush8*)(arp + kk + 8 * lhalf),
                              *(const ush8*)(arp + kk + 16 + 8 * lhalf));
        #pragma unroll
        for (int ct = 0; ct < 8; ++ct) {
            const int j = ct * 16 + lrow;
            const u16* bp = Bm + (size_t)j * 128 + kk + 8 * lhalf;
            v16bf bfr = pack_frag(*(const ush8*)bp, *(const ush8*)(bp + 16));
            acc[ct] = __builtin_amdgcn_wmma_f32_16x16x32_bf16(
                false, afr, false, bfr, (short)0, acc[ct], false, false);
        }
    }
    #pragma unroll
    for (int ct = 0; ct < 8; ++ct) {
        const int j = ct * 16 + lrow;
        const float bv = bias2[j];
        #pragma unroll
        for (int rr = 0; rr < 8; ++rr) {
            const int bb = bb0 + rr + 8 * lhalf;
            out[((size_t)bb * 128 + i) * 128 + j] = acc[ct][rr] + bv;
        }
    }
}

// ---------------------------------------------------------------------------
// Entropy top-4 mask + sigmoid-softmax gating (in place on G).
// Block = one (b, branch); wave handles 16 rows, 4 cols/lane.
// ---------------------------------------------------------------------------
__global__ __launch_bounds__(256) void k_mask(float* __restrict__ G,
                                              int*   __restrict__ sel) {
    const int b = blockIdx.x, br = blockIdx.y;
    float* Gb = G + ((size_t)br * 128 + b) * 16384;

    __shared__ float sEnt[128];
    __shared__ int   sSel[4];

    const int tid = threadIdx.x, wave = tid >> 5, lane = tid & 31;

    for (int rr = 0; rr < 16; ++rr) {
        const int i = wave * 16 + rr;
        float x[4];
        #pragma unroll
        for (int q = 0; q < 4; ++q) x[q] = Gb[i * 128 + lane + 32 * q];
        // entropy of softmax(x)
        float m = wred_max(fmaxf(fmaxf(x[0], x[1]), fmaxf(x[2], x[3])));
        float e[4], ls = 0.f;
        #pragma unroll
        for (int q = 0; q < 4; ++q) { e[q] = __expf(x[q] - m); ls += e[q]; }
        const float s = wred_sum(ls);
        float h = 0.f;
        #pragma unroll
        for (int q = 0; q < 4; ++q) { float p = e[q] / s; h += p * __logf(p + 1e-9f); }
        h = wred_sum(h);
        if (lane == 0) sEnt[i] = -h;
        // gate = softmax(sigmoid(x))
        float sg[4];
        #pragma unroll
        for (int q = 0; q < 4; ++q) sg[q] = 1.f / (1.f + __expf(-x[q]));
        float m2 = wred_max(fmaxf(fmaxf(sg[0], sg[1]), fmaxf(sg[2], sg[3])));
        float e2[4], ls2 = 0.f;
        #pragma unroll
        for (int q = 0; q < 4; ++q) { e2[q] = __expf(sg[q] - m2); ls2 += e2[q]; }
        const float s2 = wred_sum(ls2);
        #pragma unroll
        for (int q = 0; q < 4; ++q) Gb[i * 128 + lane + 32 * q] = e2[q] / s2;
    }
    __syncthreads();

    // top-4 smallest entropies (register-only reduction in wave 0)
    if (wave == 0) {
        float ev[4];
        #pragma unroll
        for (int q = 0; q < 4; ++q) ev[q] = sEnt[lane + 32 * q];
        #pragma unroll
        for (int t = 0; t < 4; ++t) {
            float lv = INFINITY; int li = 0x7fffffff;
            #pragma unroll
            for (int q = 0; q < 4; ++q) {
                const int ridx = lane + 32 * q;
                if (ev[q] < lv || (ev[q] == lv && ridx < li)) { lv = ev[q]; li = ridx; }
            }
            #pragma unroll
            for (int msk = 16; msk >= 1; msk >>= 1) {
                float ov = __shfl_xor(lv, msk, 32);
                int   oi = __shfl_xor(li, msk, 32);
                if (ov < lv || (ov == lv && oi < li)) { lv = ov; li = oi; }
            }
            if (lane == 0) { sSel[t] = li; sel[((size_t)br * 128 + b) * 4 + t] = li; }
            #pragma unroll
            for (int q = 0; q < 4; ++q) if (lane + 32 * q == li) ev[q] = INFINITY;
        }
    }
    __syncthreads();

    // zero non-selected rows
    for (int rr = 0; rr < 16; ++rr) {
        const int i = wave * 16 + rr;
        const bool keep = (i == sSel[0]) | (i == sSel[1]) | (i == sSel[2]) | (i == sSel[3]);
        if (!keep) {
            #pragma unroll
            for (int q = 0; q < 4; ++q) Gb[i * 128 + lane + 32 * q] = 0.f;
        }
    }
}

// ---------------------------------------------------------------------------
// Output zero-fill. Non-temporal (TH=NT): 268 MB of zeros must not evict
// M / gates (the sandwich kernel re-reads both) from the 192 MB L2.
// ---------------------------------------------------------------------------
__global__ void k_zero(f32x4* __restrict__ p, long n4) {
    const f32x4 z = {0.f, 0.f, 0.f, 0.f};
    for (long i = blockIdx.x * (long)blockDim.x + threadIdx.x; i < n4;
         i += (long)gridDim.x * blockDim.x)
        __builtin_nontemporal_store(z, p + i);
}

// ---------------------------------------------------------------------------
// Sparse sandwich: out[b, i_sel, d, f] = sum_{jj} tmp[i_sel, j_sel, f] * delta[j_sel, d]
//                  tmp[i,j,f] = sum_o omega[i,o] * M[b,o,j,f]
// ---------------------------------------------------------------------------
__global__ __launch_bounds__(256) void k_sandwich(const float* __restrict__ M,
                                                  const float* __restrict__ G,
                                                  const int*   __restrict__ sel,
                                                  float*       __restrict__ out) {
    const int b = blockIdx.x;
    __shared__ float sOm[4][128];
    __shared__ float sDe[4][128];
    __shared__ float sMp[128][32];
    __shared__ float sT[4][4][32];
    __shared__ int   sSo[4], sSd[4];

    const int tid = threadIdx.x;
    if (tid < 4) { sSo[tid] = sel[b * 4 + tid]; sSd[tid] = sel[(128 + b) * 4 + tid]; }
    __syncthreads();

    const float* Go = G + (size_t)b * 16384;
    const float* Gd = G + (size_t)(128 + b) * 16384;
    for (int idx = tid; idx < 512; idx += 256) {
        const int ii = idx >> 7, c = idx & 127;
        sOm[ii][c] = Go[sSo[ii] * 128 + c];
        sDe[ii][c] = Gd[sSd[ii] * 128 + c];
    }
    __syncthreads();

    const float* Mb = M + (size_t)b * 524288;
    for (int jj = 0; jj < 4; ++jj) {
        const int j = sSd[jj];
        for (int v = tid; v < 1024; v += 256) {
            const int o = v >> 3, f4 = v & 7;
            ((f32x4*)&sMp[0][0])[v] =
                *(const f32x4*)(Mb + (size_t)o * 4096 + (size_t)j * 32 + f4 * 4);
        }
        __syncthreads();
        if (tid < 128) {
            const int ii = tid >> 5, f = tid & 31;
            float s = 0.f;
            for (int o = 0; o < 128; ++o) s += sOm[ii][o] * sMp[o][f];
            sT[ii][jj][f] = s;
        }
        __syncthreads();
    }

    for (int idx = tid; idx < 16384; idx += 256) {
        const int ii = idx >> 12, d = (idx >> 5) & 127, f = idx & 31;
        const float s = sT[ii][0][f] * sDe[0][d] + sT[ii][1][f] * sDe[1][d]
                      + sT[ii][2][f] * sDe[2][d] + sT[ii][3][f] * sDe[3][d];
        out[(size_t)b * 524288 + (size_t)sSo[ii] * 4096 + (size_t)d * 32 + f] = s;
    }
}

// ---------------------------------------------------------------------------
// Launch
// ---------------------------------------------------------------------------
extern "C" void kernel_launch(void* const* d_in, const int* in_sizes, int n_in,
                              void* d_out, int out_size, void* d_ws, size_t ws_size,
                              hipStream_t stream) {
    const float* M    = (const float*)d_in[0];
    const float* Wo   = (const float*)d_in[1];
    const float* bo   = (const float*)d_in[2];
    const float* Wd   = (const float*)d_in[3];
    const float* bd   = (const float*)d_in[4];
    const float* W1o  = (const float*)d_in[5];
    const float* b_om = (const float*)d_in[6];
    const float* W1d  = (const float*)d_in[7];
    const float* b_de = (const float*)d_in[8];
    float* out = (float*)d_out;

    // workspace layout (~34 MB)
    u16*   wsW   = (u16*)d_ws;                   // [2][128][4096] bf16   (2 MB)
    u16*   wsW1  = wsW  + 2 * 524288;            // [2][128][128][128]    (8 MB)
    u16*   wsXp  = wsW1 + 2 * 2097152;           // [2][128][128][128]    (8 MB)
    float* wsG   = (float*)(wsXp + 2 * 2097152); // [2][128][128][128]    (16 MB)
    int*   wsSel = (int*)(wsG + 2 * 2097152);    // [2][128][4]

    k_cvt<<<512,  256, 0, stream>>>(Wo,  wsW,            524288);
    k_cvt<<<512,  256, 0, stream>>>(Wd,  wsW + 524288,   524288);
    k_cvt<<<2048, 256, 0, stream>>>(W1o, wsW1,           2097152);
    k_cvt<<<2048, 256, 0, stream>>>(W1d, wsW1 + 2097152, 2097152);

    k_proj    <<<dim3(128, 2), 256, 0, stream>>>(M, wsW, bo, bd, wsXp);
    k_scores  <<<dim3(128, 2), 256, 0, stream>>>(wsXp, wsW1, b_om, b_de, wsG);
    k_mask    <<<dim3(128, 2), 256, 0, stream>>>(wsG, wsSel);
    k_zero    <<<4096, 256, 0, stream>>>((f32x4*)out, 16777216L);
    k_sandwich<<<128, 256, 0, stream>>>(M, wsG, wsSel, out);
}
